// meanPoolAggregator_67757404061980
// MI455X (gfx1250) — matrix operand order repaired and checked by reading, not compiled, not run on hardware
//
#include <hip/hip_runtime.h>
#include <hip/hip_bf16.h>

// CDNA5 (gfx1250) GraphSAGE mean-pool aggregator.
//   out[n,f] = mean_s relu( sum_d emb[neighbors[n,s],d] * W[f,d] + b[f] )
//
// Tall-skinny GEMM (1.25M x 128 x 128) on v_wmma_f32_16x16x32_f16 (f32 acc).
// Embedding gather feeds A fragments straight from global (packed f32->f16
// cvt in registers). W staged once to LDS as f16; B fragments double-buffered
// in registers so ds_load_b128 of N-tile nt+1 overlaps the WMMA chain of nt.
// Mean-pool via branch-free ds_add_f32 atomics into a per-node LDS accumulator.

typedef __attribute__((ext_vector_type(16))) _Float16 v16h;
typedef __attribute__((ext_vector_type(8)))  _Float16 v8h;
typedef __attribute__((ext_vector_type(8)))  float    v8f;

#define FEAT         128
#define NUM_SAMPLE   25
#define NODES_PER_WG 16
#define ROWS_PER_WG  (NODES_PER_WG * NUM_SAMPLE)   // 400
#define MTILES       (ROWS_PER_WG / 16)            // 25
#define WPAD         (FEAT + 8)                    // padded LDS row (halfs)

// Load the 4 K-step B fragments (v16h each) for N-tile `ntv` from LDS.
#define LOAD_B(dst, ntv)                                                   \
    {                                                                      \
        const v8h* wp_ = (const v8h*)&Wl[((ntv) * 16 + m) * WPAD + kbB];   \
        _Pragma("unroll")                                                  \
        for (int ks_ = 0; ks_ < 4; ++ks_) {                                \
            union { v16h v; v8h h[2]; } ub_;                               \
            ub_.h[0] = wp_[ks_ * 4];                                       \
            ub_.h[1] = wp_[ks_ * 4 + 2];                                   \
            (dst)[ks_] = ub_.v;                                            \
        }                                                                  \
    }

__global__ __launch_bounds__(256) void
meanPoolAggregator_67757404061980_kernel(const long long* __restrict__ neighbors,
                                         const float* __restrict__ emb,
                                         const float* __restrict__ W,
                                         const float* __restrict__ bias,
                                         float* __restrict__ out,
                                         int n_nodes, int total_rows) {
    __shared__ _Float16 Wl[FEAT * WPAD];              // 34.0 KB  W as f16, padded
    __shared__ float    accum[NODES_PER_WG * FEAT];   //  8.0 KB  per-node pooled sums

    const int tid  = threadIdx.x;
    const int wave = tid >> 5;
    const int lane = tid & 31;

    // ---- stage W (f32 -> f16) into LDS; zero the accumulator ----
    for (int i = tid; i < FEAT * FEAT; i += 256) {
        int f = i >> 7, d = i & (FEAT - 1);
        Wl[f * WPAD + d] = (_Float16)W[i];
    }
    for (int i = tid; i < NODES_PER_WG * FEAT; i += 256) accum[i] = 0.0f;
    __syncthreads();

    const int m     = lane & 15;            // A row within tile (both half-waves)
    const int koffA = (lane < 16) ? 0 : 8;  // A K-half select
    const int kbB   = (lane < 16) ? 0 : 8;  // B K-half select
    const int row_wg_base = blockIdx.x * ROWS_PER_WG;
    const bool full = (row_wg_base + ROWS_PER_WG) <= total_rows;   // WG-uniform

    // bias per output column handled by this lane (one read, kept in regs)
    float bv[8];
#pragma unroll
    for (int nt = 0; nt < 8; ++nt) bv[nt] = bias[nt * 16 + m];

    for (int mtile = wave; mtile < MTILES; mtile += 8) {
        // ---------- build A fragments (4 K-steps) from the gather ----------
        int row  = row_wg_base + mtile * 16 + m;
        int rowc = (row < total_rows) ? row : (total_rows - 1);
        int idx  = (int)neighbors[rowc];
        const float* __restrict__ src = emb + idx * FEAT;

        // prefetch next M-tile's row for this wave (global_prefetch_b8)
        if (mtile + 8 < MTILES) {
            int nrow = row_wg_base + (mtile + 8) * 16 + m;
            if (nrow < total_rows) {
                int nidx = (int)neighbors[nrow];
                __builtin_prefetch(emb + nidx * FEAT + koffA, 0, 1);
            }
        }

        v16h a[4];
#pragma unroll
        for (int ks = 0; ks < 4; ++ks) {
            const float4* p = (const float4*)(src + ks * 32 + koffA);
            float4 x0 = p[0];   // K + 0..3
            float4 x1 = p[1];   // K + 4..7
            float4 x2 = p[4];   // K +16..19
            float4 x3 = p[5];   // K +20..23
            v16h va;
            va[0]  = (_Float16)x0.x; va[1]  = (_Float16)x0.y;
            va[2]  = (_Float16)x0.z; va[3]  = (_Float16)x0.w;
            va[4]  = (_Float16)x1.x; va[5]  = (_Float16)x1.y;
            va[6]  = (_Float16)x1.z; va[7]  = (_Float16)x1.w;
            va[8]  = (_Float16)x2.x; va[9]  = (_Float16)x2.y;
            va[10] = (_Float16)x2.z; va[11] = (_Float16)x2.w;
            va[12] = (_Float16)x3.x; va[13] = (_Float16)x3.y;
            va[14] = (_Float16)x3.z; va[15] = (_Float16)x3.w;
            a[ks] = va;
        }

        // ---------- per-M-tile epilogue constants (hoisted out of nt loop) ----
        const int mbase = mtile * 16 + ((lane < 16) ? 0 : 8);
        int accOff[8];
#pragma unroll
        for (int r = 0; r < 8; ++r)
            accOff[r] = ((mbase + r) / NUM_SAMPLE) * FEAT + m;   // + nt*16 per tile

        // ---------- 8 N-tiles x 4 K-step WMMAs, B double-buffered ----------
        v16h bcur[4], bnxt[4];
        LOAD_B(bcur, 0);

        if (full) {
#pragma unroll
            for (int nt = 0; nt < 8; ++nt) {
                if (nt < 7) LOAD_B(bnxt, nt + 1);    // overlap with WMMA chain
                v8f c = {};
#pragma unroll
                for (int ks = 0; ks < 4; ++ks)
                    c = __builtin_amdgcn_wmma_f32_16x16x32_f16(
                            false, a[ks], false, bcur[ks], (short)0, c, false, false);
                // bias + ReLU + pooled accumulation: branch-free ds_add_f32
#pragma unroll
                for (int r = 0; r < 8; ++r) {
                    float val = fmaxf(c[r] + bv[nt], 0.0f);
                    atomicAdd(&accum[accOff[r] + nt * 16], val);
                }
#pragma unroll
                for (int ks = 0; ks < 4; ++ks) bcur[ks] = bnxt[ks];
            }
        } else {   // tail block: per-row guard on atomics, WMMA still full-EXEC
            bool rowok[8];
#pragma unroll
            for (int r = 0; r < 8; ++r)
                rowok[r] = (row_wg_base + mbase + r) < total_rows;
#pragma unroll
            for (int nt = 0; nt < 8; ++nt) {
                if (nt < 7) LOAD_B(bnxt, nt + 1);
                v8f c = {};
#pragma unroll
                for (int ks = 0; ks < 4; ++ks)
                    c = __builtin_amdgcn_wmma_f32_16x16x32_f16(
                            false, a[ks], false, bcur[ks], (short)0, c, false, false);
#pragma unroll
                for (int r = 0; r < 8; ++r) {
                    if (rowok[r]) {
                        float val = fmaxf(c[r] + bv[nt], 0.0f);
                        atomicAdd(&accum[accOff[r] + nt * 16], val);
                    }
                }
#pragma unroll
                for (int ks = 0; ks < 4; ++ks) bcur[ks] = bnxt[ks];
            }
        }
    }

    __syncthreads();

    // ---------- mean over samples, store ----------
    const float inv_s = 1.0f / (float)NUM_SAMPLE;
    for (int i = tid; i < NODES_PER_WG * FEAT; i += 256) {
        int g = blockIdx.x * NODES_PER_WG + (i >> 7);
        if (g < n_nodes)
            out[(long long)g * FEAT + (i & (FEAT - 1))] = accum[i] * inv_s;
    }
}

extern "C" void kernel_launch(void* const* d_in, const int* in_sizes, int n_in,
                              void* d_out, int out_size, void* d_ws, size_t ws_size,
                              hipStream_t stream) {
    const long long* neighbors = (const long long*)d_in[0]; // int64 [n, 25]
    const float*     emb       = (const float*)d_in[1];     // [100000, 128]
    const float*     W         = (const float*)d_in[2];     // [128, 128]
    const float*     bias      = (const float*)d_in[3];     // [128]
    float*           out       = (float*)d_out;             // [n, 128]

    const int total_rows = in_sizes[0];                     // n * 25
    const int n_nodes = total_rows / NUM_SAMPLE;
    const int blocks  = (n_nodes + NODES_PER_WG - 1) / NODES_PER_WG;

    meanPoolAggregator_67757404061980_kernel<<<blocks, 256, 0, stream>>>(
        neighbors, emb, W, bias, out, n_nodes, total_rows);
}
